// GraphAttentionLayer_36893769073014
// MI455X (gfx1250) — compile-verified, matrix operand order checked by reading, not run on hardware
//
#include <hip/hip_runtime.h>
#include <hip/hip_bf16.h>

// ---------------------------------------------------------------------------
// Problem constants (from the reference)
// ---------------------------------------------------------------------------
#define NN     4096      // N
#define DD     2048      // D (2*D == N)
#define FOUT   512
#define ALPHA  0.2f
#define NEGV   (-9e15f)

// ---------------------------------------------------------------------------
// WMMA types (CDNA5 / gfx1250, wave32)
// ---------------------------------------------------------------------------
typedef __attribute__((ext_vector_type(16))) __bf16 v16bf;
typedef __attribute__((ext_vector_type(8)))  float  v8f;

union Frag { uint4 q[2]; unsigned int u[8]; v16bf v; };

// f32 -> bf16 round-to-nearest-even
__device__ __forceinline__ unsigned short f2b(float x) {
    unsigned int u = __float_as_uint(x);
    u += 0x7FFFu + ((u >> 16) & 1u);
    return (unsigned short)(u >> 16);
}

// ---------------------------------------------------------------------------
// Elementwise f32 -> bf16 convert
// ---------------------------------------------------------------------------
__global__ void cvt_f32_bf16(const float* __restrict__ in,
                             unsigned short* __restrict__ out, int n) {
    for (int i = blockIdx.x * 256 + threadIdx.x; i < n; i += gridDim.x * 256)
        out[i] = f2b(in[i]);
}

// ---------------------------------------------------------------------------
// bf16 x bf16 -> f32 tiled WMMA GEMM, double-buffered LDS, async A staging.
//   A: M x K   row-major bf16 (rows 16B-aligned: K % 8 == 0)
//   B: K x Nc  row-major bf16
//   C: M x Nc  f32 (written; += if accumulate)
// Block tile: 128(M) x 64(N), BK = 32. 256 threads = 8 waves (4x2 wave grid),
// each wave computes 32x32 via 2x2 v_wmma_f32_16x16x32_bf16 accumulators.
// A tile is staged with global_load_async_to_lds_b128 (ASYNCcnt);
// B tile is loaded to registers early, repacked [col][kpair], stored post-WMMA.
// ---------------------------------------------------------------------------
#define BM 128
#define BN 64
#define BK 32

__global__ __launch_bounds__(256) void gemm_bf16_wmma(
    const unsigned short* __restrict__ A,
    const unsigned short* __restrict__ B,
    float* __restrict__ C,
    int M, int Nc, int K, int accumulate)
{
    // single LDS array so async LDS byte addresses are index*4 (static base 0)
    // layout: A buf0 [0,2048) A buf1 [2048,4096) B buf0 [4096,5120) B buf1 [5120,6144)
    __shared__ unsigned int lds[6144];           // 24 KB

    const int tid  = threadIdx.x;
    const int lane = tid & 31;
    const int wid  = tid >> 5;
    const int wm   = wid & 3;        // wave row in block (0..3)
    const int wn   = wid >> 2;       // wave col in block (0..1)
    const int lsub = lane & 15;
    const int g    = lane >> 4;      // lane group

    const int blockM = blockIdx.y * BM;
    const int blockN = blockIdx.x * BN;

    const uint4* Aw4 = (const uint4*)A;
    const uint2* Bw2 = (const uint2*)B;
    const int ldA4 = K  >> 3;        // uint4 per A row
    const int ldB2 = Nc >> 2;        // uint2 per B row

    // B repack: 16 kpairs x 16 uint2-col-groups = 256 items, one per thread
    const int bkp = tid >> 4;        // k-pair 0..15
    const int bc2 = tid & 15;        // uint2 column group (4 bf16 cols)

    // A async staging: 512 uint4 per tile, 2 per thread
    const int ar0 = tid >> 2,         ac0 = tid & 3;
    const int ar1 = (tid + 256) >> 2, ac1 = tid & 3;

    v8f acc[2][2] = {};

    // ---- prologue: stage tile 0 into buffer 0 ----
    {
        const uint4* g0 = &Aw4[(size_t)(blockM + ar0) * ldA4 + ac0];
        const uint4* g1 = &Aw4[(size_t)(blockM + ar1) * ldA4 + ac1];
        unsigned l0 = (unsigned)(tid * 16);
        unsigned l1 = (unsigned)((tid + 256) * 16);
        asm volatile("global_load_async_to_lds_b128 %0, %1, off" :: "v"(l0), "v"(g0) : "memory");
        asm volatile("global_load_async_to_lds_b128 %0, %1, off" :: "v"(l1), "v"(g1) : "memory");

        uint2 d0 = Bw2[(size_t)(2 * bkp)     * ldB2 + (blockN >> 2) + bc2];
        uint2 d1 = Bw2[(size_t)(2 * bkp + 1) * ldB2 + (blockN >> 2) + bc2];
        int cbase = 4096 + (4 * bc2) * 16 + bkp;
        lds[cbase]      = (d0.x & 0xFFFFu) | (d1.x << 16);
        lds[cbase + 16] = (d0.x >> 16)     | (d1.x & 0xFFFF0000u);
        lds[cbase + 32] = (d0.y & 0xFFFFu) | (d1.y << 16);
        lds[cbase + 48] = (d0.y >> 16)     | (d1.y & 0xFFFF0000u);

        asm volatile("s_wait_asynccnt 0x0" ::: "memory");
        __syncthreads();
    }

    int cur = 0;
    for (int kt = 0; kt < K; kt += BK) {
        const int nxt = cur ^ 1;
        const bool hasNext = (kt + BK) < K;
        const int aOffCur = cur << 11;              // A buffer dword offset
        const int bOffCur = 4096 + (cur << 10);     // B buffer dword offset
        const int aOffNxt = nxt << 11;
        const int bOffNxt = 4096 + (nxt << 10);

        // ---- issue next tile's global traffic early (hides under WMMA) ----
        uint2 d0 = {0u, 0u}, d1 = {0u, 0u};
        if (hasNext) {
            const int kn = kt + BK;
            d0 = Bw2[(size_t)(kn + 2 * bkp)     * ldB2 + (blockN >> 2) + bc2];
            d1 = Bw2[(size_t)(kn + 2 * bkp + 1) * ldB2 + (blockN >> 2) + bc2];
            const uint4* g0 = &Aw4[(size_t)(blockM + ar0) * ldA4 + (kn >> 3) + ac0];
            const uint4* g1 = &Aw4[(size_t)(blockM + ar1) * ldA4 + (kn >> 3) + ac1];
            unsigned l0 = (unsigned)((aOffNxt + tid * 4) * 4);
            unsigned l1 = (unsigned)((aOffNxt + (tid + 256) * 4) * 4);
            asm volatile("global_load_async_to_lds_b128 %0, %1, off" :: "v"(l0), "v"(g0) : "memory");
            asm volatile("global_load_async_to_lds_b128 %0, %1, off" :: "v"(l1), "v"(g1) : "memory");
        }

        // ---- gather fragments (ISA 16-bit A 16x32 lane/VGPR layout) ----
        // lane group g: VGPR j in [0,3] -> kpair g*4+j ; j in [4,7] -> kpair 8+g*4+(j-4)
        // 16B-aligned contiguous -> two ds_load_b128 per fragment
        Frag a[2], b[2];
        #pragma unroll
        for (int f = 0; f < 2; ++f) {
            int row   = wm * 32 + f * 16 + lsub;
            int abase = aOffCur + row * 16 + g * 4;
            a[f].q[0] = *(const uint4*)&lds[abase];
            a[f].q[1] = *(const uint4*)&lds[abase + 8];
            int col   = wn * 32 + f * 16 + lsub;
            int bbase = bOffCur + col * 16 + g * 4;
            b[f].q[0] = *(const uint4*)&lds[bbase];
            b[f].q[1] = *(const uint4*)&lds[bbase + 8];
        }

        #pragma unroll
        for (int fa = 0; fa < 2; ++fa)
            #pragma unroll
            for (int fb = 0; fb < 2; ++fb)
                acc[fa][fb] = __builtin_amdgcn_wmma_f32_16x16x32_bf16(
                    false, a[fa].v, false, b[fb].v,
                    (short)0, acc[fa][fb], false, false);

        // ---- repack + store next B tile (loadcnt wait lands after WMMAs) ----
        if (hasNext) {
            int cbase = bOffNxt + (4 * bc2) * 16 + bkp;
            lds[cbase]      = (d0.x & 0xFFFFu) | (d1.x << 16);
            lds[cbase + 16] = (d0.x >> 16)     | (d1.x & 0xFFFF0000u);
            lds[cbase + 32] = (d0.y & 0xFFFFu) | (d1.y << 16);
            lds[cbase + 48] = (d0.y >> 16)     | (d1.y & 0xFFFF0000u);
        }

        asm volatile("s_wait_asynccnt 0x0" ::: "memory");
        __syncthreads();
        cur = nxt;
    }

    // ---- epilogue: C frag layout: VGPR r -> row r + g*8, col = lane%16 ----
    #pragma unroll
    for (int fa = 0; fa < 2; ++fa) {
        #pragma unroll
        for (int fb = 0; fb < 2; ++fb) {
            #pragma unroll
            for (int r = 0; r < 8; ++r) {
                int grow = blockM + wm * 32 + fa * 16 + g * 8 + r;
                int gcol = blockN + wn * 32 + fb * 16 + lsub;
                size_t idx = (size_t)grow * Nc + gcol;
                float val = acc[fa][fb][r];
                if (accumulate) val += C[idx];
                C[idx] = val;
            }
        }
    }
}

// ---------------------------------------------------------------------------
// f1 = h @ a1, f2 = h @ a2   (wave per row, wave32 shuffle reduction)
// ---------------------------------------------------------------------------
__global__ void f12_kernel(const float* __restrict__ h, const float* __restrict__ a,
                           float* __restrict__ f1, float* __restrict__ f2) {
    int wid = threadIdx.x >> 5, lane = threadIdx.x & 31;
    int row = blockIdx.x * 8 + wid;
    float s1 = 0.f, s2 = 0.f;
    for (int f = lane; f < FOUT; f += 32) {
        float hv = h[(size_t)row * FOUT + f];
        s1 += hv * a[f];
        s2 += hv * a[FOUT + f];
    }
    #pragma unroll
    for (int o = 16; o > 0; o >>= 1) {
        s1 += __shfl_xor(s1, o, 32);
        s2 += __shfl_xor(s2, o, 32);
    }
    if (lane == 0) { f1[row] = s1; f2[row] = s2; }
}

// ---------------------------------------------------------------------------
// Row softmax over masked leaky-relu logits; emits bf16 exp weights + row sum.
// One block (256 threads) per row.
// ---------------------------------------------------------------------------
__global__ __launch_bounds__(256) void softmax_kernel(
    const float* __restrict__ f1, const float* __restrict__ f2,
    const float* __restrict__ sLeft, const float* __restrict__ feat,
    unsigned short* __restrict__ attB, float* __restrict__ rowsum)
{
    __shared__ float red[256];
    const int i = blockIdx.x, tid = threadIdx.x;
    const float f1i = f1[i];

    // pass 1: row max of masked logits
    float m = NEGV;
    for (int j = tid; j < NN; j += 256) {
        float sv = (j < DD) ? sLeft[(size_t)i * DD + j]
                            : feat[(size_t)i * DD + (j - DD)];
        float v = NEGV;
        if (sv > 0.f) {
            float t = f1i + f2[j];
            v = (t > 0.f) ? t : ALPHA * t;
        }
        m = fmaxf(m, v);
    }
    red[tid] = m; __syncthreads();
    for (int s = 128; s > 0; s >>= 1) {
        if (tid < s) red[tid] = fmaxf(red[tid], red[tid + s]);
        __syncthreads();
    }
    const float rm = red[0];
    __syncthreads();

    // pass 2: exp + sum, store bf16 weights
    float sum = 0.f;
    for (int j = tid; j < NN; j += 256) {
        float sv = (j < DD) ? sLeft[(size_t)i * DD + j]
                            : feat[(size_t)i * DD + (j - DD)];
        float v = NEGV;
        if (sv > 0.f) {
            float t = f1i + f2[j];
            v = (t > 0.f) ? t : ALPHA * t;
        }
        float p = __expf(v - rm);
        attB[(size_t)i * NN + j] = f2b(p);
        sum += p;
    }
    red[tid] = sum; __syncthreads();
    for (int s = 128; s > 0; s >>= 1) {
        if (tid < s) red[tid] += red[tid + s];
        __syncthreads();
    }
    if (tid == 0) rowsum[i] = red[0];
}

// ---------------------------------------------------------------------------
// out = elu(hp / rowsum)
// ---------------------------------------------------------------------------
__global__ void finalize_kernel(const float* __restrict__ hp,
                                const float* __restrict__ rowsum,
                                float* __restrict__ out, int n) {
    for (int idx = blockIdx.x * 256 + threadIdx.x; idx < n; idx += gridDim.x * 256) {
        int i = idx >> 9;   // / FOUT (512)
        float x = hp[idx] / rowsum[i];
        out[idx] = (x > 0.f) ? x : (__expf(x) - 1.f);
    }
}

// ---------------------------------------------------------------------------
// Host-side orchestration
// ---------------------------------------------------------------------------
extern "C" void kernel_launch(void* const* d_in, const int* in_sizes, int n_in,
                              void* d_out, int out_size, void* d_ws, size_t ws_size,
                              hipStream_t stream) {
    const float* features = (const float*)d_in[0];   // (4096, 2048)
    const float* support  = (const float*)d_in[1];   // (4096, 4096)
    const float* W        = (const float*)d_in[2];   // (4096, 512)
    const float* a        = (const float*)d_in[3];   // (1024, 1)
    float* out            = (float*)d_out;           // (4096, 512)

    char* ws = (char*)d_ws;
    // workspace layout (bytes). attExpB aliases supB (dead after GEMM1).
    const size_t offFeatB  = 0;                                // 16 MiB
    const size_t offSupB   = offFeatB  + (size_t)NN * DD * 2;  // 32 MiB
    const size_t offSLeft  = offSupB   + (size_t)NN * NN * 2;  // f32 32 MiB
    const size_t offSLeftB = offSLeft  + (size_t)NN * DD * 4;  // 16 MiB
    const size_t offWb     = offSLeftB + (size_t)NN * DD * 2;  // 4 MiB
    const size_t offH      = offWb     + (size_t)NN * FOUT * 2;// f32 8 MiB
    const size_t offHb     = offH      + (size_t)NN * FOUT * 4;// 4 MiB
    const size_t offF12    = offHb     + (size_t)NN * FOUT * 2;
    const size_t offRowS   = offF12    + (size_t)2 * NN * 4;
    const size_t offHP     = offRowS   + (size_t)NN * 4;       // f32 8 MiB

    unsigned short* featB  = (unsigned short*)(ws + offFeatB);
    unsigned short* supB   = (unsigned short*)(ws + offSupB);
    unsigned short* attB   = supB;                 // alias: supB dead after GEMM1
    float*          sLeft  = (float*)(ws + offSLeft);
    unsigned short* sLeftB = (unsigned short*)(ws + offSLeftB);
    unsigned short* Wb     = (unsigned short*)(ws + offWb);
    float*          h      = (float*)(ws + offH);
    unsigned short* hB     = (unsigned short*)(ws + offHb);
    float*          f1     = (float*)(ws + offF12);
    float*          f2     = f1 + NN;
    float*          rowsum = (float*)(ws + offRowS);
    float*          hp     = (float*)(ws + offHP);

    // 1. downconvert inputs to bf16
    cvt_f32_bf16<<<(NN * DD) / 256, 256, 0, stream>>>(features, featB, NN * DD);
    cvt_f32_bf16<<<(NN * NN) / 256, 256, 0, stream>>>(support, supB, NN * NN);
    cvt_f32_bf16<<<(NN * FOUT) / 256, 256, 0, stream>>>(W, Wb, NN * FOUT);

    // 2. sLeft = support @ features   (M=4096, Nc=2048, K=4096)
    gemm_bf16_wmma<<<dim3(DD / BN, NN / BM), 256, 0, stream>>>(
        supB, featB, sLeft, NN, DD, NN, 0);

    // 3. h = [sLeft | features] @ W   (split K at 2048, accumulate)
    cvt_f32_bf16<<<(NN * DD) / 256, 256, 0, stream>>>(sLeft, sLeftB, NN * DD);
    gemm_bf16_wmma<<<dim3(FOUT / BN, NN / BM), 256, 0, stream>>>(
        sLeftB, Wb, h, NN, FOUT, DD, 0);
    gemm_bf16_wmma<<<dim3(FOUT / BN, NN / BM), 256, 0, stream>>>(
        featB, Wb + (size_t)DD * FOUT, h, NN, FOUT, DD, 1);

    // 4. attention logit projections
    f12_kernel<<<NN / 8, 256, 0, stream>>>(h, a, f1, f2);

    // 5. masked softmax -> bf16 exp weights + row sums
    softmax_kernel<<<NN, 256, 0, stream>>>(f1, f2, sLeft, features, attB, rowsum);

    // 6. hp = attExp @ h
    cvt_f32_bf16<<<(NN * FOUT) / 256, 256, 0, stream>>>(h, hB, NN * FOUT);
    gemm_bf16_wmma<<<dim3(FOUT / BN, NN / BM), 256, 0, stream>>>(
        attB, hB, hp, NN, FOUT, NN, 0);

    // 7. out = elu(hp / rowsum)
    finalize_kernel<<<(NN * FOUT) / 256, 256, 0, stream>>>(hp, rowsum, out, NN * FOUT);
}
